// TrajGenerator_90074054132125
// MI455X (gfx1250) — matrix operand order repaired and checked by reading, not compiled, not run on hardware
//
#include <hip/hip_runtime.h>
#include <math.h>

// ---------------------------------------------------------------------------
// TrajGenerator on MI455X (gfx1250): bf16 WMMA GEMMs + fp32 recurrence state.
// Dims: HL=20, T=32, B=2048, hidden=128, gates=384, OP=30.
// GEMM: 256-thread block = 8 waves -> 128x64 C tile; B tile staged in LDS
// (double-buffered, fragment-ordered), A streamed from global, WMMA bf16->f32.
// ---------------------------------------------------------------------------

#define HL 20
#define TT 32
#define BB 2048
#define HID 128
#define GG 384
#define CE 32
#define ENCIN 96
#define DECIN 256
#define OPL 30

typedef __attribute__((ext_vector_type(16))) __bf16        v16bf;
typedef __attribute__((ext_vector_type(8)))  float         v8f;
typedef __attribute__((ext_vector_type(8)))  unsigned int  v8u;

__device__ __forceinline__ unsigned short f2bf(float f) {
    unsigned int u = __builtin_bit_cast(unsigned int, f);
    u += 0x7fffu + ((u >> 16) & 1u);          // round-to-nearest-even
    return (unsigned short)(u >> 16);
}
__device__ __forceinline__ float sigm(float x) { return 1.0f / (1.0f + expf(-x)); }

// ---------------------------------------------------------------------------
// C[M,N] = A[M,K](bf16,row-major,lda) * Bt[N,K](bf16 pre-transposed,ldb) + bias
// Block: 256 threads = 8 waves. Block tile: 128 rows x 64 cols (wave w -> rows
// m0+16w). Per K-step (32): B tile 64x32 bf16 (4KB) cooperatively loaded into
// LDS in WMMA-fragment order; each lane then reads 32 contiguous bytes per
// fragment. Double-buffered; next tile's global loads overlap current WMMAs.
// Requires M%128==0, N%64==0, K%32==0 (all shapes here satisfy this).
// ---------------------------------------------------------------------------
__global__ void __launch_bounds__(256)
k_gemm_bf16(const unsigned short* __restrict__ A, int lda,
            const unsigned short* __restrict__ Bt, int ldb,
            const float* __restrict__ bias,
            float* __restrict__ C, int ldc,
            int M, int N, int K) {
    __shared__ unsigned short bs[2][64 * 32];     // 2 x 4KB, fragment-ordered

    const int tid  = threadIdx.x;
    const int lane = tid & 31;
    const int wid  = tid >> 5;
    const int ntn  = N >> 6;
    const int mt   = blockIdx.x / ntn;
    const int nt   = blockIdx.x - mt * ntn;
    const int m0   = (mt << 7) + (wid << 4);      // this wave's 16 rows
    const int n0   = nt << 6;

    // ---- cooperative B loader: thread tid owns 8 ushorts (one 16B chunk) ----
    // flat element f = tid*8 -> n = tid/4 (0..63), k = (tid%4)*8 (0,8,16,24)
    const int ln = tid >> 2;
    const int lk = (tid & 3) << 3;
    const unsigned short* bsrc = Bt + (size_t)(n0 + ln) * ldb + lk;
    // LDS offset in fragment order: element (n,k) lives at
    //   ((n/16)*32 + (k/16)*16 + n%16)*16 + (k%16)
    const int ldst = (((ln >> 4) << 5) + ((lk >> 4) << 4) + (ln & 15)) * 16 + (lk & 15);

    // ---- A fragment addressing (contiguous k = 16*(lane/16) + s, same order as B in LDS) ----
    const int am  = m0 + (lane & 15);
    const int akb = (lane >> 4) << 4;             // 0 or 16
    const unsigned short* arow = A + (size_t)am * lda;

    v8f acc[4];
#pragma unroll
    for (int c = 0; c < 4; ++c)
#pragma unroll
        for (int r = 0; r < 8; ++r) acc[c][r] = 0.0f;

#define LOAD_A(dst, k0)                                                     \
    {                                                                       \
        _Pragma("unroll")                                                   \
        for (int v = 0; v < 4; ++v) {                                       \
            dst[v]     = *(const unsigned int*)(arow + (k0) + akb + 2 * v); \
            dst[v + 4] = *(const unsigned int*)(arow + (k0) + 8 + akb + 2 * v); \
        }                                                                   \
    }

    // prologue: tile 0 -> LDS buf 0, A frag 0 -> regs
    {
        uint4 b0 = *(const uint4*)bsrc;
        *(uint4*)(&bs[0][ldst]) = b0;
    }
    v8u aCur; LOAD_A(aCur, 0);
    __syncthreads();

    for (int k0 = 0; k0 < K; k0 += 32) {
        const int  buf  = (k0 >> 5) & 1;
        const bool more = (k0 + 32) < K;
        uint4 nb; v8u aNxt;
        if (more) {                                // overlap with WMMAs below
            nb = *(const uint4*)(bsrc + (k0 + 32));
            LOAD_A(aNxt, k0 + 32);
        }

        // read 4 B fragments from LDS (32 contiguous bytes each)
        v8u bu[4];
#pragma unroll
        for (int c = 0; c < 4; ++c) {
            const uint4* p = (const uint4*)(&bs[buf][((c << 5) + lane) << 4]);
            uint4 x0 = p[0], x1 = p[1];
            bu[c][0] = x0.x; bu[c][1] = x0.y; bu[c][2] = x0.z; bu[c][3] = x0.w;
            bu[c][4] = x1.x; bu[c][5] = x1.y; bu[c][6] = x1.z; bu[c][7] = x1.w;
        }
        v16bf af = __builtin_bit_cast(v16bf, aCur);
#pragma unroll
        for (int c = 0; c < 4; ++c) {
            acc[c] = __builtin_amdgcn_wmma_f32_16x16x32_bf16(
                false, af, false, __builtin_bit_cast(v16bf, bu[c]),
                (short)0, acc[c], false, false);
        }

        if (more) {
            *(uint4*)(&bs[buf ^ 1][ldst]) = nb;
            aCur = aNxt;
        }
        __syncthreads();
    }

    // C/D layout: VGPR r -> M=m0+r(+8 upper half), N = n0+c*16+(lane&15)
#pragma unroll
    for (int c = 0; c < 4; ++c) {
        const int   n  = n0 + (c << 4) + (lane & 15);
        const float bv = bias ? bias[n] : 0.0f;
#pragma unroll
        for (int r = 0; r < 8; ++r) {
            const int m = m0 + r + ((lane >> 4) << 3);
            C[(size_t)m * ldc + n] = acc[c][r] + bv;
        }
    }
#undef LOAD_A
}

// --------------------------- elementwise kernels ---------------------------

__global__ void k_w2bf_t(const float* __restrict__ W, unsigned short* __restrict__ Wt,
                         int K, int N) {   // W[K,N] -> Wt[N,K] bf16
    int i = blockIdx.x * blockDim.x + threadIdx.x;
    if (i >= K * N) return;
    int k = i / N, n = i - k * N;
    Wt[(size_t)n * K + k] = f2bf(W[i]);
}

__global__ void k_zero_h(float* __restrict__ h, unsigned short* __restrict__ hb, int n) {
    int i = blockIdx.x * blockDim.x + threadIdx.x;
    if (i >= n) return;
    h[i] = 0.0f; hb[i] = 0;
}

__global__ void k_lengths(const float* __restrict__ waypts, int* __restrict__ len) {
    int b = blockIdx.x * blockDim.x + threadIdx.x;
    if (b >= BB) return;
    int c = 1;
    for (int t = 1; t < TT; ++t) {
        float s = fabsf(waypts[((size_t)t * BB + b) * 2]) +
                  fabsf(waypts[((size_t)t * BB + b) * 2 + 1]);
        c += (s != 0.0f) ? 1 : 0;
    }
    len[b] = c;
}

__global__ void k_hist_embed(const float* __restrict__ hist,
                             const float* __restrict__ w, const float* __restrict__ bias,
                             unsigned short* __restrict__ xbf, int total) {
    int i = blockIdx.x * blockDim.x + threadIdx.x;
    if (i >= total) return;
    int row = i >> 5, ce = i & 31;
    float s = bias[ce];
    for (int j = 0; j < 5; ++j) s += hist[(size_t)row * 5 + j] * w[j * CE + ce];
    s = s > 0.0f ? s : 0.1f * s;
    xbf[i] = f2bf(s);
}

__global__ void k_waypt_embed(const float* __restrict__ waypts,
                              const float* __restrict__ scene,
                              const float* __restrict__ agent,
                              const float* __restrict__ cw, const float* __restrict__ cb,
                              const float* __restrict__ sw, const float* __restrict__ sb,
                              const float* __restrict__ aw, const float* __restrict__ ab,
                              unsigned short* __restrict__ emb, int total) {
    int i = blockIdx.x * blockDim.x + threadIdx.x;
    if (i >= total) return;
    int row = i / ENCIN, o = i - row * ENCIN;
    float s;
    if (o < 32) {
        s = cb[o];
        for (int j = 0; j < 2; ++j) s += waypts[(size_t)row * 2 + j] * cw[j * 32 + o];
    } else if (o < 64) {
        int oo = o - 32; s = sb[oo];
        for (int j = 0; j < 32; ++j) s += scene[(size_t)row * 32 + j] * sw[j * 32 + oo];
    } else {
        int oo = o - 64; s = ab[oo];
        for (int j = 0; j < 32; ++j) s += agent[(size_t)row * 32 + j] * aw[j * 32 + oo];
    }
    s = s > 0.0f ? s : 0.1f * s;
    emb[i] = f2bf(s);
}

__global__ void k_reverse_emb(const unsigned short* __restrict__ emb,
                              const int* __restrict__ len,
                              unsigned short* __restrict__ er, int total) {
    int i = blockIdx.x * blockDim.x + threadIdx.x;
    if (i >= total) return;
    int t = i / (BB * ENCIN), rem = i - t * BB * ENCIN;
    int b = rem / ENCIN, o = rem - b * ENCIN;
    int it = len[b] - 1 - t;
    it = it < 0 ? 0 : (it > TT - 1 ? TT - 1 : it);
    er[i] = emb[((size_t)it * BB + b) * ENCIN + o];
}

// GRU gate math. gi/gh already include their biases (added in GEMM epilogue).
__global__ void k_gru_gate(const float* __restrict__ gi, const float* __restrict__ gh,
                           float* __restrict__ h, unsigned short* __restrict__ hbf) {
    int i = blockIdx.x * blockDim.x + threadIdx.x;
    if (i >= BB * HID) return;
    int b = i >> 7, j = i & 127;
    const float* gib = gi + (size_t)b * GG;
    const float* ghb = gh + (size_t)b * GG;
    float r  = sigm(gib[j] + ghb[j]);
    float z  = sigm(gib[HID + j] + ghb[HID + j]);
    float nn = tanhf(gib[2 * HID + j] + r * ghb[2 * HID + j]);
    float hp = h[i];
    float hn = (1.0f - z) * nn + z * hp;
    h[i] = hn; hbf[i] = f2bf(hn);
}

__global__ void k_gru_gate_masked(const float* __restrict__ gi, const float* __restrict__ gh,
                                  float* __restrict__ h, unsigned short* __restrict__ hbf,
                                  float* __restrict__ out, const int* __restrict__ len, int t) {
    int i = blockIdx.x * blockDim.x + threadIdx.x;
    if (i >= BB * HID) return;
    int b = i >> 7, j = i & 127;
    const float* gib = gi + (size_t)b * GG;
    const float* ghb = gh + (size_t)b * GG;
    float r  = sigm(gib[j] + ghb[j]);
    float z  = sigm(gib[HID + j] + ghb[HID + j]);
    float nn = tanhf(gib[2 * HID + j] + r * ghb[2 * HID + j]);
    float hp = h[i];
    float hn = (1.0f - z) * nn + z * hp;
    bool mt = t < len[b];
    float hs = mt ? hn : hp;
    h[i] = hs; hbf[i] = f2bf(hs);
    out[i] = mt ? hs : 0.0f;
}

__global__ void k_build_hwaypt(const float* __restrict__ outf,
                               const float* __restrict__ outbr,
                               const int* __restrict__ len,
                               float* __restrict__ hwf, unsigned short* __restrict__ hwb,
                               int total) {
    int i = blockIdx.x * blockDim.x + threadIdx.x;
    if (i >= total) return;
    int t = i / (BB * DECIN), rem = i - t * BB * DECIN;
    int b = rem >> 8, c = rem & 255;
    float v;
    if (c < HID) {
        v = outf[((size_t)t * BB + b) * HID + c];
    } else {
        int l = len[b];
        if (t < l) {
            int it = l - 1 - t;
            v = outbr[((size_t)it * BB + b) * HID + (c - HID)];
        } else v = 0.0f;
    }
    hwf[i] = v; hwb[i] = f2bf(v);
}

__global__ void k_attn_scores(const float* __restrict__ hW, const float* __restrict__ wpW,
                              const float* __restrict__ w2, const float* __restrict__ b2,
                              float* __restrict__ e) {
    int i = blockIdx.x * blockDim.x + threadIdx.x;   // i = t*B + b
    if (i >= TT * BB) return;
    int b = i & (BB - 1);
    const float* hr = hW  + (size_t)b * 128;
    const float* wr = wpW + (size_t)i * 128;
    float s = b2[0];
    for (int n = 0; n < 128; ++n) s += tanhf(hr[n] + wr[n]) * w2[n];
    e[i] = s;
}

__global__ void k_softmax_t(const float* __restrict__ e, float* __restrict__ a) {
    int b = blockIdx.x * blockDim.x + threadIdx.x;
    if (b >= BB) return;
    float mx = -3.0e38f;
    for (int t = 0; t < TT; ++t) { float v = e[(size_t)t * BB + b]; mx = v > mx ? v : mx; }
    float sum = 0.0f;
    for (int t = 0; t < TT; ++t) sum += expf(e[(size_t)t * BB + b] - mx);
    float inv = 1.0f / sum;
    for (int t = 0; t < TT; ++t) a[(size_t)t * BB + b] = expf(e[(size_t)t * BB + b] - mx) * inv;
}

__global__ void k_ip(const float* __restrict__ a, const float* __restrict__ hwf,
                     unsigned short* __restrict__ ipbf) {
    int i = blockIdx.x * blockDim.x + threadIdx.x;   // i = b*256 + c
    if (i >= BB * DECIN) return;
    int b = i >> 8, c = i & 255;
    float s = 0.0f;
    for (int t = 0; t < TT; ++t)
        s += a[(size_t)t * BB + b] * hwf[((size_t)t * BB + b) * DECIN + c];
    ipbf[i] = f2bf(s);
}

__global__ void k_proj_out(const float* __restrict__ h, const float* __restrict__ opw,
                           const float* __restrict__ opb, float* __restrict__ out, int s) {
    int i = blockIdx.x * blockDim.x + threadIdx.x;   // i = b*2 + k
    if (i >= BB * 2) return;
    int b = i >> 1, k = i & 1;
    float v = opb[k];
    for (int j = 0; j < HID; ++j) v += h[(size_t)b * HID + j] * opw[j * 2 + k];
    out[(size_t)b * (OPL * 2) + s * 2 + k] = v;
}

// ---------------------------------------------------------------------------

extern "C" void kernel_launch(void* const* d_in, const int* in_sizes, int n_in,
                              void* d_out, int out_size, void* d_ws, size_t ws_size,
                              hipStream_t stream) {
    (void)in_sizes; (void)n_in; (void)out_size; (void)ws_size;
    const float* hist      = (const float*)d_in[0];
    const float* waypts    = (const float*)d_in[1];
    const float* scene     = (const float*)d_in[2];
    const float* agent     = (const float*)d_in[3];
    const float* hist_w    = (const float*)d_in[4];
    const float* hist_b    = (const float*)d_in[5];
    const float* gh_wi     = (const float*)d_in[6];
    const float* gh_wh     = (const float*)d_in[7];
    const float* gh_bi     = (const float*)d_in[8];
    const float* gh_bh     = (const float*)d_in[9];
    const float* coord_w   = (const float*)d_in[10];
    const float* coord_b   = (const float*)d_in[11];
    const float* scene_w   = (const float*)d_in[12];
    const float* scene_b   = (const float*)d_in[13];
    const float* agent_w   = (const float*)d_in[14];
    const float* agent_b   = (const float*)d_in[15];
    const float* gf_wi     = (const float*)d_in[16];
    const float* gf_wh     = (const float*)d_in[17];
    const float* gf_bi     = (const float*)d_in[18];
    const float* gf_bh     = (const float*)d_in[19];
    const float* gb_wi     = (const float*)d_in[20];
    const float* gb_wh     = (const float*)d_in[21];
    const float* gb_bi     = (const float*)d_in[22];
    const float* gb_bh     = (const float*)d_in[23];
    const float* dec_wi    = (const float*)d_in[24];
    const float* dec_wh    = (const float*)d_in[25];
    const float* dec_bi    = (const float*)d_in[26];
    const float* dec_bh    = (const float*)d_in[27];
    const float* attn1_w   = (const float*)d_in[28];
    const float* attn1_b   = (const float*)d_in[29];
    const float* attn2_w   = (const float*)d_in[30];
    const float* attn2_b   = (const float*)d_in[31];
    const float* op_w      = (const float*)d_in[32];
    const float* op_b      = (const float*)d_in[33];
    float* out = (float*)d_out;

    // ---- workspace bump allocator (256B aligned) ----
    char* ws = (char*)d_ws;
    size_t off = 0;
    auto alloc = [&](size_t bytes) -> void* {
        void* p = ws + off;
        off += (bytes + 255) & ~(size_t)255;
        return p;
    };
    auto U = [&](size_t n) { return (unsigned short*)alloc(n * 2); };
    auto F = [&](size_t n) { return (float*)alloc(n * 4); };

    unsigned short* wt_gh_wi  = U((size_t)GG * CE);
    unsigned short* wt_gh_wh  = U((size_t)GG * HID);
    unsigned short* wt_gf_wi  = U((size_t)GG * ENCIN);
    unsigned short* wt_gf_wh  = U((size_t)GG * HID);
    unsigned short* wt_gb_wi  = U((size_t)GG * ENCIN);
    unsigned short* wt_gb_wh  = U((size_t)GG * HID);
    unsigned short* wt_dec_wi = U((size_t)GG * DECIN);
    unsigned short* wt_dec_wh = U((size_t)GG * HID);
    unsigned short* wt_attn1  = U((size_t)128 * GG);     // [128][384]
    int*            lengths   = (int*)alloc((size_t)BB * 4);
    unsigned short* xhist_bf  = U((size_t)HL * BB * CE);
    float*          gi_hist   = F((size_t)HL * BB * GG);
    float*          gh_scr    = F((size_t)BB * GG);
    float*          h_dec     = F((size_t)BB * HID);     // hist-encoder -> decoder state
    unsigned short* h_dec_bf  = U((size_t)BB * HID);
    unsigned short* emb_bf    = U((size_t)TT * BB * ENCIN);
    unsigned short* embr_bf   = U((size_t)TT * BB * ENCIN);
    float*          gi_wp     = F((size_t)TT * BB * GG); // reused fwd then bwd
    float*          h_wp      = F((size_t)BB * HID);
    unsigned short* h_wp_bf   = U((size_t)BB * HID);
    float*          out_f     = F((size_t)TT * BB * HID);
    float*          out_br    = F((size_t)TT * BB * HID);
    float*          hw_f32    = F((size_t)TT * BB * DECIN);
    unsigned short* hw_bf     = U((size_t)TT * BB * DECIN);
    float*          wpW       = F((size_t)TT * BB * 128);
    float*          hW        = F((size_t)BB * 128);
    float*          e_scr     = F((size_t)TT * BB);
    float*          a_scr     = F((size_t)TT * BB);
    unsigned short* ip_bf     = U((size_t)BB * DECIN);
    float*          gi_dec    = F((size_t)BB * GG);

    auto grd = [](long n) { return dim3((unsigned)((n + 255) / 256)); };
    auto gemm = [&](const unsigned short* A, int lda, const unsigned short* Bt, int ldb,
                    const float* bias, float* C, int ldc, int M, int N, int K) {
        int blocks = (M >> 7) * (N >> 6);                // exact: M%128==0, N%64==0
        k_gemm_bf16<<<blocks, 256, 0, stream>>>(A, lda, Bt, ldb, bias, C, ldc, M, N, K);
    };
    auto tr = [&](const float* W, unsigned short* Wt, int K, int N) {
        k_w2bf_t<<<grd((long)K * N), 256, 0, stream>>>(W, Wt, K, N);
    };

    // ---- weight prep: fp32 [K,N] -> bf16 [N,K] ----
    tr(gh_wi, wt_gh_wi, CE, GG);     tr(gh_wh, wt_gh_wh, HID, GG);
    tr(gf_wi, wt_gf_wi, ENCIN, GG);  tr(gf_wh, wt_gf_wh, HID, GG);
    tr(gb_wi, wt_gb_wi, ENCIN, GG);  tr(gb_wh, wt_gb_wh, HID, GG);
    tr(dec_wi, wt_dec_wi, DECIN, GG);tr(dec_wh, wt_dec_wh, HID, GG);
    tr(attn1_w, wt_attn1, GG, 128);  // rows 0..127 = h-part (K off 0), 128..383 = wp-part

    k_lengths<<<grd(BB), 256, 0, stream>>>(waypts, lengths);

    // ---- history encoder: batched gi, 20 serial gh+gate steps ----
    k_hist_embed<<<grd((long)HL * BB * CE), 256, 0, stream>>>(hist, hist_w, hist_b,
                                                              xhist_bf, HL * BB * CE);
    gemm(xhist_bf, CE, wt_gh_wi, CE, gh_bi, gi_hist, GG, HL * BB, GG, CE);
    k_zero_h<<<grd((long)BB * HID), 256, 0, stream>>>(h_dec, h_dec_bf, BB * HID);
    for (int t = 0; t < HL; ++t) {
        gemm(h_dec_bf, HID, wt_gh_wh, HID, gh_bh, gh_scr, GG, BB, GG, HID);
        k_gru_gate<<<grd((long)BB * HID), 256, 0, stream>>>(
            gi_hist + (size_t)t * BB * GG, gh_scr, h_dec, h_dec_bf);
    }

    // ---- waypoint embedding + reversed copy ----
    k_waypt_embed<<<grd((long)TT * BB * ENCIN), 256, 0, stream>>>(
        waypts, scene, agent, coord_w, coord_b, scene_w, scene_b, agent_w, agent_b,
        emb_bf, TT * BB * ENCIN);
    k_reverse_emb<<<grd((long)TT * BB * ENCIN), 256, 0, stream>>>(
        emb_bf, lengths, embr_bf, TT * BB * ENCIN);

    // ---- forward masked GRU ----
    gemm(emb_bf, ENCIN, wt_gf_wi, ENCIN, gf_bi, gi_wp, GG, TT * BB, GG, ENCIN);
    k_zero_h<<<grd((long)BB * HID), 256, 0, stream>>>(h_wp, h_wp_bf, BB * HID);
    for (int t = 0; t < TT; ++t) {
        gemm(h_wp_bf, HID, wt_gf_wh, HID, gf_bh, gh_scr, GG, BB, GG, HID);
        k_gru_gate_masked<<<grd((long)BB * HID), 256, 0, stream>>>(
            gi_wp + (size_t)t * BB * GG, gh_scr, h_wp, h_wp_bf,
            out_f + (size_t)t * BB * HID, lengths, t);
    }
    // ---- backward masked GRU (gi buffer reused) ----
    gemm(embr_bf, ENCIN, wt_gb_wi, ENCIN, gb_bi, gi_wp, GG, TT * BB, GG, ENCIN);
    k_zero_h<<<grd((long)BB * HID), 256, 0, stream>>>(h_wp, h_wp_bf, BB * HID);
    for (int t = 0; t < TT; ++t) {
        gemm(h_wp_bf, HID, wt_gb_wh, HID, gb_bh, gh_scr, GG, BB, GG, HID);
        k_gru_gate_masked<<<grd((long)BB * HID), 256, 0, stream>>>(
            gi_wp + (size_t)t * BB * GG, gh_scr, h_wp, h_wp_bf,
            out_br + (size_t)t * BB * HID, lengths, t);
    }
    k_build_hwaypt<<<grd((long)TT * BB * DECIN), 256, 0, stream>>>(
        out_f, out_br, lengths, hw_f32, hw_bf, TT * BB * DECIN);

    // ---- attention precompute: wpW = h_waypt @ attn1_w[128:384] + attn1_b ----
    gemm(hw_bf, DECIN, wt_attn1 + 128, GG, attn1_b, wpW, 128, TT * BB, 128, DECIN);

    // ---- decoder: 30 serial steps ----
    for (int s = 0; s < OPL; ++s) {
        gemm(h_dec_bf, HID, wt_attn1, GG, nullptr, hW, 128, BB, 128, HID);
        k_attn_scores<<<grd((long)TT * BB), 256, 0, stream>>>(hW, wpW, attn2_w, attn2_b, e_scr);
        k_softmax_t<<<grd(BB), 256, 0, stream>>>(e_scr, a_scr);
        k_ip<<<grd((long)BB * DECIN), 256, 0, stream>>>(a_scr, hw_f32, ip_bf);
        gemm(ip_bf, DECIN, wt_dec_wi, DECIN, dec_bi, gi_dec, GG, BB, GG, DECIN);
        gemm(h_dec_bf, HID, wt_dec_wh, HID, dec_bh, gh_scr, GG, BB, GG, HID);
        k_gru_gate<<<grd((long)BB * HID), 256, 0, stream>>>(gi_dec, gh_scr, h_dec, h_dec_bf);
        k_proj_out<<<grd((long)BB * 2), 256, 0, stream>>>(h_dec, op_w, op_b, out, s);
    }
}
